// LLTM_15195594293829
// MI455X (gfx1250) — matrix operand order repaired and checked by reading, not compiled, not run on hardware
//
#include <hip/hip_runtime.h>

typedef __bf16 bf16;
typedef __attribute__((ext_vector_type(16))) __bf16 v16bf;
typedef __attribute__((ext_vector_type(8)))  float  v8f;
typedef __attribute__((ext_vector_type(4)))  int    v4i;

constexpr int S_DIM = 1024;          // state size
constexpr int K_DIM = 2048;          // IN + S
constexpr int B_DIM = 4096;          // batch
constexpr int BM = 128;              // block tile M (batch)
constexpr int BN = 64;               // block tile N (state columns, per gate)
constexpr int BK = 64;               // K tile (2 WMMA K-steps)
constexpr int LDK = BK + 8;          // padded LDS row stride (72 bf16 = 144B)

constexpr int A_ELEMS = BM * LDK;        // 9216 bf16  (18 KB)
constexpr int B_ELEMS = 4 * BN * LDK;    // 18432 bf16 (36 KB)
constexpr int BUF_ELEMS = A_ELEMS + B_ELEMS;
constexpr size_t LDS_BYTES = (size_t)2 * BUF_ELEMS * sizeof(bf16);  // 108 KB

#if __has_builtin(__builtin_amdgcn_global_load_async_to_lds_b128) && \
    __has_builtin(__builtin_amdgcn_s_wait_asynccnt)
#define USE_ASYNC 1
#else
#define USE_ASYNC 0
#endif

union FragBF {
    v16bf v;
    uint4 q[2];
};

union Pack4 {
    bf16 h[4];
    uint2 u;
};

#if USE_ASYNC
typedef __attribute__((address_space(1))) v4i* as1_v4i_ptr;
typedef __attribute__((address_space(3))) v4i* as3_v4i_ptr;

__device__ __forceinline__ void async_cp16(const bf16* g, bf16* l) {
    // generic->AS1 / generic->AS3 via integer round-trip (global generic addr ==
    // global addr; LDS generic addr low 32 bits == LDS byte offset)
    as1_v4i_ptr gp = (as1_v4i_ptr)(uintptr_t)g;
    as3_v4i_ptr lp = (as3_v4i_ptr)(uint32_t)(uintptr_t)l;
    __builtin_amdgcn_global_load_async_to_lds_b128(gp, lp, 0, 0);
}
#endif

// ---- fp32 -> bf16 conversion: Xb = [input | old_h], row-major [4096, 2048] ----
__global__ __launch_bounds__(256) void cvt_x_kernel(const float* __restrict__ in,
                                                    const float* __restrict__ oldh,
                                                    bf16* __restrict__ Xb) {
    int gid = blockIdx.x * blockDim.x + threadIdx.x;   // one thread = 4 elements
    int idx = gid * 4;
    int row = idx >> 11;            // / 2048
    int col = idx & 2047;
    const float* src = (col < S_DIM) ? (in + (size_t)row * S_DIM + col)
                                     : (oldh + (size_t)row * S_DIM + (col - S_DIM));
    float4 f = *(const float4*)src;
    Pack4 p;
    p.h[0] = (bf16)f.x; p.h[1] = (bf16)f.y; p.h[2] = (bf16)f.z; p.h[3] = (bf16)f.w;
    *(uint2*)(Xb + idx) = p.u;
}

// ---- fp32 -> bf16 conversion: Wb row-major [4096, 2048] ----
__global__ __launch_bounds__(256) void cvt_w_kernel(const float* __restrict__ w,
                                                    bf16* __restrict__ Wb) {
    int gid = blockIdx.x * blockDim.x + threadIdx.x;
    int idx = gid * 4;
    float4 f = *(const float4*)(w + idx);
    Pack4 p;
    p.h[0] = (bf16)f.x; p.h[1] = (bf16)f.y; p.h[2] = (bf16)f.z; p.h[3] = (bf16)f.w;
    *(uint2*)(Wb + idx) = p.u;
}

// ---- fused 4-gate GEMM + LSTM cell update ----
__global__ __launch_bounds__(256) void lltm_gemm_kernel(
    const bf16* __restrict__ Xb,        // [4096, 2048]
    const bf16* __restrict__ Wb,        // [4096, 2048] (f, i, o, g row blocks)
    const float* __restrict__ bias,     // [4S]
    const float* __restrict__ old_cell, // [B, S]
    float* __restrict__ new_h,          // [B, S]
    float* __restrict__ new_cell)       // [B, S]
{
    extern __shared__ bf16 smem[];      // 2 x (A tile + B tiles)

    const int tid  = threadIdx.x;
    const int lane = tid & 31;
    const int wave = tid >> 5;          // 0..7
    const int wm   = wave & 3;          // wave row (M): 0..3, 32 rows each
    const int wn   = wave >> 2;         // wave col (N): 0..1, 32 cols each
    const int m0   = blockIdx.x * BM;   // batch base
    const int s0   = blockIdx.y * BN;   // state-column base

    const int aRow[4] = { (tid + 0*256) >> 3, (tid + 1*256) >> 3,
                          (tid + 2*256) >> 3, (tid + 3*256) >> 3 };
    const int aCol[4] = { ((tid + 0*256) & 7) * 8, ((tid + 1*256) & 7) * 8,
                          ((tid + 2*256) & 7) * 8, ((tid + 3*256) & 7) * 8 };

    v8f acc[2][2][4];
    #pragma unroll
    for (int tm = 0; tm < 2; ++tm)
        #pragma unroll
        for (int tn = 0; tn < 2; ++tn)
            #pragma unroll
            for (int g = 0; g < 4; ++g)
                acc[tm][tn][g] = (v8f){0.f, 0.f, 0.f, 0.f, 0.f, 0.f, 0.f, 0.f};

    constexpr int T = K_DIM / BK;       // 32 K-tiles

    auto compute_tile = [&](int buf) {
        const bf16* Asb = smem + buf * BUF_ELEMS;
        const bf16* Bsb = Asb + A_ELEMS;
        #pragma unroll
        for (int kk = 0; kk < 2; ++kk) {
            const int kb    = kk * 32;
            const int aHalf = (lane < 16) ? 0 : 8;   // ISA A-layout K split
            const int bHalf = (lane < 16) ? 0 : 8;   // ISA B-layout K split

            FragBF a[2];
            #pragma unroll
            for (int tm = 0; tm < 2; ++tm) {
                const bf16* p = &Asb[(wm * 32 + tm * 16 + (lane & 15)) * LDK + kb + aHalf];
                a[tm].q[0] = *(const uint4*)p;        // K = half .. half+7
                a[tm].q[1] = *(const uint4*)(p + 16); // K = half+16 .. half+23
            }
            #pragma unroll
            for (int g = 0; g < 4; ++g) {
                #pragma unroll
                for (int tn = 0; tn < 2; ++tn) {
                    FragBF b;
                    const bf16* p =
                        &Bsb[(g * BN + wn * 32 + tn * 16 + (lane & 15)) * LDK + kb + bHalf];
                    b.q[0] = *(const uint4*)p;        // K = half .. half+7
                    b.q[1] = *(const uint4*)(p + 16); // K = half+16 .. half+23
                    #pragma unroll
                    for (int tm = 0; tm < 2; ++tm) {
                        acc[tm][tn][g] = __builtin_amdgcn_wmma_f32_16x16x32_bf16(
                            false, a[tm].v, false, b.v,
                            (short)0, acc[tm][tn][g], false, false);
                    }
                }
            }
        }
    };

#if USE_ASYNC
    // -------- async ping-pong pipeline: no staging VGPRs, one barrier/tile ----
    auto async_tile = [&](int k0, int buf) {
        bf16* Asb = smem + buf * BUF_ELEMS;
        bf16* Bsb = Asb + A_ELEMS;
        #pragma unroll
        for (int i = 0; i < 4; ++i)
            async_cp16(&Xb[(size_t)(m0 + aRow[i]) * K_DIM + k0 + aCol[i]],
                       &Asb[aRow[i] * LDK + aCol[i]]);
        #pragma unroll
        for (int g = 0; g < 4; ++g) {
            #pragma unroll
            for (int i = 0; i < 2; ++i) {
                int chunk = tid + i * 256;        // 0..511 within a 64x64 gate tile
                int r = chunk >> 3;
                int c = (chunk & 7) * 8;
                async_cp16(&Wb[(size_t)(g * S_DIM + s0 + r) * K_DIM + k0 + c],
                           &Bsb[(g * BN + r) * LDK + c]);
            }
        }
    };

    async_tile(0, 0);
    for (int t = 0; t < T; ++t) {
        __builtin_amdgcn_s_wait_asynccnt(0);  // this wave's tile-t copies landed
        __syncthreads();                      // every wave's copies landed
        if (t + 1 < T)
            async_tile((t + 1) * BK, (t + 1) & 1);  // overlaps compute below
        compute_tile(t & 1);
    }
#else
    // -------- fallback: synchronous single-buffer staging (round-1 style) ----
    for (int t = 0; t < T; ++t) {
        const int k0 = t * BK;
        bf16* Asb = smem;
        bf16* Bsb = smem + A_ELEMS;
        #pragma unroll
        for (int i = 0; i < 4; ++i)
            *(uint4*)&Asb[aRow[i] * LDK + aCol[i]] =
                *(const uint4*)&Xb[(size_t)(m0 + aRow[i]) * K_DIM + k0 + aCol[i]];
        #pragma unroll
        for (int g = 0; g < 4; ++g) {
            #pragma unroll
            for (int i = 0; i < 2; ++i) {
                int chunk = tid + i * 256;
                int r = chunk >> 3;
                int c = (chunk & 7) * 8;
                *(uint4*)&Bsb[(g * BN + r) * LDK + c] =
                    *(const uint4*)&Wb[(size_t)(g * S_DIM + s0 + r) * K_DIM + k0 + c];
            }
        }
        __syncthreads();
        compute_tile(0);
        __syncthreads();
    }
#endif

    // ---- fused LSTM epilogue on register accumulators ----
    #pragma unroll
    for (int tm = 0; tm < 2; ++tm) {
        #pragma unroll
        for (int tn = 0; tn < 2; ++tn) {
            const int scol = s0 + wn * 32 + tn * 16 + (lane & 15);
            const float bf_ = bias[scol];
            const float bi_ = bias[scol + S_DIM];
            const float bo_ = bias[scol + 2 * S_DIM];
            const float bg_ = bias[scol + 3 * S_DIM];
            #pragma unroll
            for (int j = 0; j < 8; ++j) {
                const int brow = m0 + wm * 32 + tm * 16 + ((lane < 16) ? j : j + 8);
                const float xf = acc[tm][tn][0][j] + bf_;
                const float xi = acc[tm][tn][1][j] + bi_;
                const float xo = acc[tm][tn][2][j] + bo_;
                const float xg = acc[tm][tn][3][j] + bg_;
                const float f = 1.f / (1.f + __expf(-xf));
                const float i = 1.f / (1.f + __expf(-xi));
                const float o = 1.f / (1.f + __expf(-xo));
                const float g = 2.f / (1.f + __expf(-2.f * xg)) - 1.f;   // tanh
                const size_t off = (size_t)brow * S_DIM + scol;
                const float c = f * old_cell[off] + i * g;
                const float t2 = 2.f / (1.f + __expf(-2.f * c)) - 1.f;   // tanh
                new_h[off]    = o * t2;
                new_cell[off] = c;
            }
        }
    }
}

extern "C" void kernel_launch(void* const* d_in, const int* in_sizes, int n_in,
                              void* d_out, int out_size, void* d_ws, size_t ws_size,
                              hipStream_t stream) {
    const float* input    = (const float*)d_in[0];   // [4096, 1024]
    const float* old_h    = (const float*)d_in[1];   // [4096, 1024]
    const float* old_cell = (const float*)d_in[2];   // [4096, 1024]
    const float* weights  = (const float*)d_in[3];   // [4096, 2048]
    const float* bias     = (const float*)d_in[4];   // [4096]

    float* out      = (float*)d_out;                 // new_h then new_cell
    float* new_h    = out;
    float* new_cell = out + (size_t)B_DIM * S_DIM;

    char* ws = (char*)d_ws;
    bf16* Xb = (bf16*)ws;                                          // 16 MB
    bf16* Wb = (bf16*)(ws + (size_t)B_DIM * K_DIM * sizeof(bf16)); // +16 MB

    const int cvt_threads = 256;
    const int n_elems     = B_DIM * K_DIM;                         // 8M elements
    cvt_x_kernel<<<n_elems / 4 / cvt_threads, cvt_threads, 0, stream>>>(input, old_h, Xb);
    cvt_w_kernel<<<n_elems / 4 / cvt_threads, cvt_threads, 0, stream>>>(weights, Wb);

    dim3 grid(B_DIM / BM, S_DIM / BN);                             // 32 x 16
    lltm_gemm_kernel<<<grid, 256, LDS_BYTES, stream>>>(Xb, Wb, bias, old_cell,
                                                       new_h, new_cell);
}
